// LinformerAttention_19834158973062
// MI455X (gfx1250) — matrix operand order repaired and checked by reading, not compiled, not run on hardware
//
#include <hip/hip_runtime.h>
#include <hip/hip_bf16.h>

typedef __attribute__((ext_vector_type(16))) __bf16 v16bf;
typedef __attribute__((ext_vector_type(8)))  __bf16 v8bf;
typedef __attribute__((ext_vector_type(2)))  __bf16 v2bf;
typedef __attribute__((ext_vector_type(8)))  float  v8f;

#define BH_ 64
#define N_  4096
#define D_  64
#define M_  256

#define LDS_NP 40   // padded n-stride of LDS tiles: 80B rows -> b128 reads conflict-free

static __device__ inline v16bf cat8(v8bf lo, v8bf hi) {
  return __builtin_shufflevector(lo, hi, 0, 1, 2, 3, 4, 5, 6, 7,
                                 8, 9, 10, 11, 12, 13, 14, 15);
}

// Convert two float4 chunks into 8 bf16 lanes of a v16bf at [base, base+8)
static __device__ inline void cvt8(v16bf& d, int base, float4 lo, float4 hi) {
  d[base + 0] = (__bf16)lo.x; d[base + 1] = (__bf16)lo.y;
  d[base + 2] = (__bf16)lo.z; d[base + 3] = (__bf16)lo.w;
  d[base + 4] = (__bf16)hi.x; d[base + 5] = (__bf16)hi.y;
  d[base + 6] = (__bf16)hi.z; d[base + 7] = (__bf16)hi.w;
}

// Pack values from two consecutive n-rows into one 4B LDS store (v_cvt_pk path)
static __device__ inline void stpair(__bf16* p, float a, float b) {
  v2bf t; t[0] = (__bf16)a; t[1] = (__bf16)b;
  *reinterpret_cast<v2bf*>(p) = t;
}

static __device__ inline float4 ldf4(const float* p) {
  return *reinterpret_cast<const float4*>(p);
}

// ---------------------------------------------------------------------------
// Kernel 1: K_proj = E^T x K, V_proj = E^T x V (per head), bf16 outputs.
//   Kp  stored [M, D] row-major   (stage-2 B tiles contiguous in d)
//   VpT stored [D, M] row-major   (stage-3 B tiles contiguous in m)
// All 8 waves of a block share one head; each 32x64 f32 chunk of K and V is
// staged ONCE through LDS (coalesced float4 loads -> transposed bf16 tile
// Kt[d][n]), software-pipelined one chunk ahead. All 8 B tiles are preloaded
// into distinct registers so the 8 WMMAs issue back-to-back without dscnt
// stalls. One wave per 16-row m-tile computes all 4 d-tiles.
// ---------------------------------------------------------------------------
__global__ __launch_bounds__(256)
void linformer_proj_kernel(const float* __restrict__ K,
                           const float* __restrict__ V,
                           const float* __restrict__ E,
                           __bf16* __restrict__ Kp,
                           __bf16* __restrict__ VpT) {
  __shared__ __align__(16) __bf16 KVs[2][D_][LDS_NP];  // [0]=Kt[d][n], [1]=Vt[d][n]

  const int tid  = threadIdx.x;
  const int wave = tid >> 5;
  const int lane = tid & 31;
  const int half = lane >> 4;
  const int l16  = lane & 15;
  const int bh   = blockIdx.x >> 1;
  const int mt   = ((blockIdx.x & 1) << 3) + wave;   // 0..15
  const int m0   = mt * 16;

  // Staging role: lanes 0-15 of a wave load one full 256B row (coalesced),
  // lanes 16-31 the next row; row-pairs packed into b32 LDS stores.
  const int rp = tid >> 4;   // row-pair selector: rows 2*rp, 2*rp+1 (0..15)
  const int cq = tid & 15;   // float4 column quad (0..15)

  const float* __restrict__ Kb = K + (size_t)bh * N_ * D_;
  const float* __restrict__ Vb = V + (size_t)bh * N_ * D_;

  v8f accK[4] = {};
  v8f accV[4] = {};

  // ---- Prologue: fetch chunk 0 into staging registers ----
  float4 k0, k1, w0, w1;
  {
    const size_t r0 = (size_t)(2 * rp) * D_ + cq * 4;
    k0 = ldf4(Kb + r0); k1 = ldf4(Kb + r0 + D_);
    w0 = ldf4(Vb + r0); w1 = ldf4(Vb + r0 + D_);
  }

  for (int n0 = 0; n0 < N_; n0 += 32) {
    // ---- Commit staged chunk to LDS (pairwise bf16 pack, b32 stores) ----
    {
      const int c = cq * 4, n = 2 * rp;
      stpair(&KVs[0][c + 0][n], k0.x, k1.x);
      stpair(&KVs[0][c + 1][n], k0.y, k1.y);
      stpair(&KVs[0][c + 2][n], k0.z, k1.z);
      stpair(&KVs[0][c + 3][n], k0.w, k1.w);
      stpair(&KVs[1][c + 0][n], w0.x, w1.x);
      stpair(&KVs[1][c + 1][n], w0.y, w1.y);
      stpair(&KVs[1][c + 2][n], w0.z, w1.z);
      stpair(&KVs[1][c + 3][n], w0.w, w1.w);
    }
    __syncthreads();

    // ---- Prefetch next chunk (global, overlaps with WMMA burst below) ----
    if (n0 + 32 < N_) {
      const size_t r0 = (size_t)(n0 + 32 + 2 * rp) * D_ + cq * 4;
      k0 = ldf4(Kb + r0); k1 = ldf4(Kb + r0 + D_);
      w0 = ldf4(Vb + r0); w1 = ldf4(Vb + r0 + D_);
    }

    // ---- A-matrix (16x32 bf16) from E: row=l16, k = half*16+e ----
    v16bf a;
#pragma unroll
    for (int e = 0; e < 16; ++e) {
      const int ka = half * 16 + e;
      a[e] = (__bf16)E[(size_t)(n0 + ka) * M_ + m0 + l16];
    }

    // ---- Preload ALL 8 B tiles into distinct registers (one ds clause) ----
    v16bf bk[4], bv[4];
#pragma unroll
    for (int dt = 0; dt < 4; ++dt) {
      // B element e of lane L = chunk[k = half*16 + e][d = dt*16 + l16]
      const __bf16* kt = &KVs[0][dt * 16 + l16][half * 16];
      const __bf16* vt = &KVs[1][dt * 16 + l16][half * 16];
      bk[dt] = cat8(*reinterpret_cast<const v8bf*>(kt),
                    *reinterpret_cast<const v8bf*>(kt + 8));
      bv[dt] = cat8(*reinterpret_cast<const v8bf*>(vt),
                    *reinterpret_cast<const v8bf*>(vt + 8));
    }

    // ---- 8 back-to-back WMMAs ----
#pragma unroll
    for (int dt = 0; dt < 4; ++dt) {
      accK[dt] = __builtin_amdgcn_wmma_f32_16x16x32_bf16(
          false, a, false, bk[dt], (short)0, accK[dt], false, false);
      accV[dt] = __builtin_amdgcn_wmma_f32_16x16x32_bf16(
          false, a, false, bv[dt], (short)0, accV[dt], false, false);
    }
    __syncthreads();
  }

  __bf16* __restrict__ Kpb = Kp  + (size_t)bh * M_ * D_;
  __bf16* __restrict__ Vpb = VpT + (size_t)bh * D_ * M_;
#pragma unroll
  for (int dt = 0; dt < 4; ++dt) {
#pragma unroll
    for (int v = 0; v < 8; ++v) {
      // C layout: row = m0 + (half ? v+8 : v), col = dt*16 + l16
      const int row = m0 + (half ? v + 8 : v);
      const int col = dt * 16 + l16;
      Kpb[(size_t)row * D_ + col] = (__bf16)accK[dt][v];
      Vpb[(size_t)col * M_ + row] = (__bf16)accV[dt][v];
    }
  }
}

// ---------------------------------------------------------------------------
// Kernel 2: attention. 8 waves/block, each wave = 16 query rows x all M=256.
// Scores register-resident (16 C tiles); softmax via shfl_xor butterflies
// within 16-lane halves; LDS only to re-layout exp(S) into WMMA-A format.
// All bf16 tile loads are explicit 16B vector loads (b128).
// ---------------------------------------------------------------------------
__global__ __launch_bounds__(256)
void linformer_attn_kernel(const float* __restrict__ Q,
                           const __bf16* __restrict__ Kp,
                           const __bf16* __restrict__ VpT,
                           float* __restrict__ Out) {
  __shared__ __align__(16) __bf16 Plds[8][16][M_ + 16];  // row stride 544B, 16B-aligned

  const int wave = threadIdx.x >> 5;
  const int lane = threadIdx.x & 31;
  const int half = lane >> 4;
  const int l16  = lane & 15;
  const int bh   = blockIdx.x >> 5;                 // 32 q-blocks per head
  const int q0   = (blockIdx.x & 31) * 128 + wave * 16;

  const float*  __restrict__ Qb  = Q   + ((size_t)bh * N_ + q0) * D_;
  const __bf16* __restrict__ Kpb = Kp  + (size_t)bh * M_ * D_;
  const __bf16* __restrict__ Vpb = VpT + (size_t)bh * D_ * M_;

  // ---- Load Q A-tiles once (k = d, two chunks of 32), float4 vector loads ----
  v16bf qa0, qa1;
  {
    const float4* qp =
        reinterpret_cast<const float4*>(Qb + (size_t)l16 * D_ + half * 8);
    cvt8(qa0, 0, qp[0],  qp[1]);    // k = half*8 + 0..7
    cvt8(qa0, 8, qp[4],  qp[5]);    // k = half*8 + 16..23
    cvt8(qa1, 0, qp[8],  qp[9]);    // k = 32 + half*8 + 0..7
    cvt8(qa1, 8, qp[12], qp[13]);   // k = 32 + half*8 + 16..23
  }

  // ---- Stage 2: S = Q x Kp^T, 16 register-resident C tiles ----
  v8f s[16];
#pragma unroll
  for (int mt = 0; mt < 16; ++mt) {
    const v8bf* kv = reinterpret_cast<const v8bf*>(
        Kpb + (size_t)(mt * 16 + l16) * D_ + (half << 3));
    const v16bf b0 = cat8(kv[0], kv[2]);   // d = half*8 + 0..7, half*8 + 16..23
    const v16bf b1 = cat8(kv[4], kv[6]);   // d = 32 + half*8 + 0..7, 32 + half*8 + 16..23
    v8f c = {};
    c = __builtin_amdgcn_wmma_f32_16x16x32_bf16(false, qa0, false, b0,
                                                (short)0, c, false, false);
    c = __builtin_amdgcn_wmma_f32_16x16x32_bf16(false, qa1, false, b1,
                                                (short)0, c, false, false);
    s[mt] = c;
  }

  // ---- Softmax: row max/sum via per-lane partials + shfl_xor(1,2,4,8) ----
  const float scale = 0.125f;  // 1/sqrt(64)
  float rmax[8];
#pragma unroll
  for (int v = 0; v < 8; ++v) {
    float m = s[0][v];
#pragma unroll
    for (int mt = 1; mt < 16; ++mt) m = fmaxf(m, s[mt][v]);
#pragma unroll
    for (int off = 1; off <= 8; off <<= 1)
      m = fmaxf(m, __shfl_xor(m, off, 32));
    rmax[v] = m * scale;
  }

  float rsum[8];
#pragma unroll
  for (int v = 0; v < 8; ++v) rsum[v] = 0.0f;
#pragma unroll
  for (int mt = 0; mt < 16; ++mt) {
#pragma unroll
    for (int v = 0; v < 8; ++v) {
      const float p = __expf(s[mt][v] * scale - rmax[v]);
      rsum[v] += p;
      // C-layout element -> P[row][m] in LDS (A-layout source for stage 3)
      Plds[wave][half ? v + 8 : v][mt * 16 + l16] = (__bf16)p;
    }
  }

  float invs[8];
#pragma unroll
  for (int v = 0; v < 8; ++v) {
    float t = rsum[v];
#pragma unroll
    for (int off = 1; off <= 8; off <<= 1)
      t += __shfl_xor(t, off, 32);
    invs[v] = 1.0f / t;
  }

  // ---- Stage 3: O = P x Vp, normalize by 1/sum at the store ----
#pragma unroll
  for (int dt = 0; dt < 4; ++dt) {
    v8f o = {};
#pragma unroll
    for (int mk = 0; mk < 8; ++mk) {       // M=256 in k-chunks of 32
      const int mbase = mk * 32;
      // A from LDS: two 16B ds_load_b128 per lane
      const __bf16* ap = &Plds[wave][l16][mbase + half * 8];
      const v16bf a = cat8(*reinterpret_cast<const v8bf*>(ap),
                           *reinterpret_cast<const v8bf*>(ap + 16));
      // B from VpT: two 16B chunks per lane (k = half*8 + 0..7, half*8 + 16..23)
      const v8bf* bp = reinterpret_cast<const v8bf*>(
          Vpb + (size_t)(dt * 16 + l16) * M_ + mbase + (half << 3));
      const v16bf b = cat8(bp[0], bp[2]);
      o = __builtin_amdgcn_wmma_f32_16x16x32_bf16(false, a, false, b,
                                                  (short)0, o, false, false);
    }
    float* __restrict__ ob = Out + ((size_t)bh * N_ + q0) * D_ + dt * 16 + l16;
#pragma unroll
    for (int v = 0; v < 8; ++v) {
      const int row = half ? v + 8 : v;
      ob[(size_t)row * D_] = o[v] * invs[v];
    }
  }
}

// ---------------------------------------------------------------------------
extern "C" void kernel_launch(void* const* d_in, const int* in_sizes, int n_in,
                              void* d_out, int out_size, void* d_ws, size_t ws_size,
                              hipStream_t stream) {
  const float* Q = (const float*)d_in[0];
  const float* K = (const float*)d_in[1];
  const float* V = (const float*)d_in[2];
  const float* E = (const float*)d_in[3];

  // Workspace: Kp bf16 [BH][M][D] (2 MB) + VpT bf16 [BH][D][M] (2 MB)
  __bf16* Kp  = (__bf16*)d_ws;
  __bf16* VpT = Kp + (size_t)BH_ * M_ * D_;

  linformer_proj_kernel<<<BH_ * 2, 256, 0, stream>>>(K, V, E, Kp, VpT);
  linformer_attn_kernel<<<BH_ * 32, 256, 0, stream>>>(Q, Kp, VpT, (float*)d_out);
}
